// CNN_59863254171658
// MI455X (gfx1250) — matrix operand order repaired and checked by reading, not compiled, not run on hardware
//
#include <hip/hip_runtime.h>
#include <hip/hip_bf16.h>
#include <math.h>

typedef _Float16 v16h __attribute__((ext_vector_type(16)));
typedef _Float16 v8h  __attribute__((ext_vector_type(8)));
typedef float    v8f  __attribute__((ext_vector_type(8)));

#define B_TOT 2048
#define OHW   10
#define P_POS 100
#define KDIM  400
#define KT    13      // 416 / 32 WMMA K-steps
#define NT    5       // 80 / 16 N tiles (32 untied | 32 parallel | 1 gate | pad)
#define NPAD  80
#define LDA   424     // padded LDS row stride (halves)
#define MT    64      // batch rows per block

// ---------------- Kernel 1: conv1(5x5,pad2) + bias + relu + maxpool2 -> h16 [B][14][14][16] f16 (channels-last)
__global__ __launch_bounds__(256)
void k_conv1(const float* __restrict__ x, const float* __restrict__ w,
             const float* __restrict__ bias, _Float16* __restrict__ h16) {
  __shared__ float sw[16 * 25];
  __shared__ float sb[16];
  for (int i = threadIdx.x; i < 400; i += blockDim.x) sw[i] = w[i];
  if (threadIdx.x < 16) sb[threadIdx.x] = bias[threadIdx.x];
  __syncthreads();
  int id = blockIdx.x * blockDim.x + threadIdx.x;
  if (id >= B_TOT * 14 * 14) return;
  int b = id / 196; int r = id % 196; int y14 = r / 14; int x14 = r % 14;
  const float* xb = x + (size_t)b * 784;
  float win[6][6];
  int y0 = 2 * y14 - 2, x0 = 2 * x14 - 2;
  #pragma unroll
  for (int i = 0; i < 6; i++)
    #pragma unroll
    for (int j = 0; j < 6; j++) {
      int yy = y0 + i, xx = x0 + j;
      win[i][j] = (yy >= 0 && yy < 28 && xx >= 0 && xx < 28) ? xb[yy * 28 + xx] : 0.0f;
    }
  _Float16* outp = h16 + (size_t)id * 16;
  for (int c = 0; c < 16; c++) {
    float s00 = 0.f, s01 = 0.f, s10 = 0.f, s11 = 0.f;
    #pragma unroll
    for (int ky = 0; ky < 5; ky++)
      #pragma unroll
      for (int kx = 0; kx < 5; kx++) {
        float wv = sw[c * 25 + ky * 5 + kx];
        s00 = fmaf(win[ky][kx],     wv, s00);
        s01 = fmaf(win[ky][kx + 1], wv, s01);
        s10 = fmaf(win[ky + 1][kx], wv, s10);
        s11 = fmaf(win[ky + 1][kx + 1], wv, s11);
      }
    float m = fmaxf(fmaxf(s00, s01), fmaxf(s10, s11)) + sb[c];
    m = fmaxf(m, 0.0f);   // relu-then-maxpool == maxpool-then-relu (monotone)
    outp[c] = (_Float16)m;
  }
}

// ---------------- Kernel 0: fuse uc_w/pc_w/wl_w (+biases as K-row 400) into per-position
// B matrices, f32->f16, K permuted to k'=spatial*16+channel, pre-swizzled into the CDNA5
// B-fragment lane layout: wfrag[p][kt][nt][lane][16]
__global__ __launch_bounds__(256)
void k_wswz(const float* __restrict__ ucw, const float* __restrict__ ucb,
            const float* __restrict__ pcw, const float* __restrict__ pcb,
            const float* __restrict__ wlw, const float* __restrict__ wlb,
            _Float16* __restrict__ wfrag) {
  const int per_p = KT * NT * 512;
  int id = blockIdx.x * blockDim.x + threadIdx.x;
  if (id >= P_POS * per_p) return;
  int p  = id / per_p;  int r  = id % per_p;
  int kt = r / (NT * 512); int r2 = r % (NT * 512);
  int nt = r2 / 512;       int r3 = r2 % 512;
  int lane = r3 >> 4;      int e  = r3 & 15;
  int n = nt * 16 + (lane & 15);
  int k = kt * 32 + (e < 8 ? (lane >> 4) * 8 + e : 16 + (lane >> 4) * 8 + (e - 8));   // B-matrix 32x16: lane=col, K in table order (8h+e | 16+8h+e-8)
  float v = 0.0f;
  if (k < KDIM) {
    int s = k >> 4, c = k & 15;             // k' = s*16 + c
    int korig = c * 25 + s;                 // torch-unfold order: channel-major
    if (n < 32)       v = ucw[((size_t)(p * KDIM + korig)) * 32 + n];
    else if (n < 64)  v = pcw[((n - 32) * 16 + c) * 25 + s];
    else if (n == 64) v = wlw[korig];
  } else if (k == KDIM) {                   // bias row (A supplies 1.0 here)
    if (n < 32)       v = ucb[n * P_POS + p];
    else if (n < 64)  v = pcb[n - 32];
    else if (n == 64) v = wlb[0];
  }
  wfrag[id] = (_Float16)v;
}

// Multi-accumulator tile worker: one A fragment feeds NCNT WMMAs per K-step.
template<int NCNT>
__device__ __forceinline__
void wave_tiles(const _Float16* __restrict__ lA, const _Float16* __restrict__ WF,
                int msub, int n0, int lane, float* __restrict__ G, int b0, int p) {
  int khalf = lane >> 4;
  int lrow  = lane & 15;
  v8f acc[NCNT] = {};
  #pragma unroll
  for (int kt = 0; kt < KT; kt++) {
    // A fragment (16-bit A 16x32 layout): two 16B LDS reads, reused across NCNT tiles
    const _Float16* pa = lA + (msub * 16 + lrow) * LDA + kt * 32 + khalf * 8;
    v8h a0 = *(const v8h*)pa;
    v8h a1 = *(const v8h*)(pa + 16);
    v16h av = __builtin_shufflevector(a0, a1, 0,1,2,3,4,5,6,7,8,9,10,11,12,13,14,15);
    #pragma unroll
    for (int j = 0; j < NCNT; j++) {
      const _Float16* pb = WF + (((size_t)kt * NT + (n0 + j)) * 32 + lane) * 16;
      v8h b0v = *(const v8h*)pb;
      v8h b1v = *(const v8h*)(pb + 8);
      v16h bv = __builtin_shufflevector(b0v, b1v, 0,1,2,3,4,5,6,7,8,9,10,11,12,13,14,15);
      acc[j] = __builtin_amdgcn_wmma_f32_16x16x32_f16(false, av, false, bv, (short)0, acc[j], false, false);
    }
  }
  // D layout: lane(0-15)=N, vgpr r -> M = khalf*8 + r
  #pragma unroll
  for (int j = 0; j < NCNT; j++)
    #pragma unroll
    for (int rr = 0; rr < 8; rr++) {
      int m = msub * 16 + khalf * 8 + rr;
      G[((size_t)(b0 + m) * P_POS + p) * NPAD + (n0 + j) * 16 + lrow] = acc[j][rr];
    }
}

// ---------------- Kernel 2: fused untied+parallel+gate GEMM.
// Per block: one position p, 64 batch rows. A (unfold patches + bias row) staged in LDS,
// B fragments streamed from L2-resident pre-swizzled weights.
// Waves 0-3: msub=w, nsubs {0,1,2}; waves 4-7: msub=w-4, nsubs {3,4}.
__global__ __launch_bounds__(256)
void k_gemm(const _Float16* __restrict__ h16, const _Float16* __restrict__ wfrag,
            float* __restrict__ G) {
  __shared__ _Float16 lA[MT * LDA];
  int p  = blockIdx.y;
  int b0 = blockIdx.x * MT;
  int oh = p / OHW, ow = p % OHW;
  int tid = threadIdx.x;

  // stage A: 25 spatial taps x 16 channels (32B contiguous) per batch row
  for (int seg = tid; seg < MT * 25; seg += 256) {
    int rrow = seg / 25, s = seg % 25;
    int kh = s / 5, kw = s % 5;
    const uint4* src = (const uint4*)(h16 + (((size_t)(b0 + rrow) * 14 + oh + kh) * 14 + (ow + kw)) * 16);
    uint4* dst = (uint4*)(lA + rrow * LDA + s * 16);
    dst[0] = src[0];
    dst[1] = src[1];
  }
  // bias row (k=400 -> 1.0) + zero pad (401..423)
  for (int i = tid; i < MT * (LDA - KDIM); i += 256) {
    int rrow = i / (LDA - KDIM); int c = i % (LDA - KDIM);
    lA[rrow * LDA + KDIM + c] = (c == 0) ? (_Float16)1.0f : (_Float16)0.0f;
  }
  __syncthreads();

  int wave = tid >> 5;
  int lane = tid & 31;
  int msub = wave & 3;
  const _Float16* WF = wfrag + (size_t)p * (KT * NT * 512);
  if (wave < 4) wave_tiles<3>(lA, WF, msub, 0, lane, G, b0, p);
  else          wave_tiles<2>(lA, WF, msub, 3, lane, G, b0, p);
}

// ---------------- Kernel 3: sigmoid gate mix + relu + maxpool2 -> flat [B,800]
__global__ __launch_bounds__(256)
void k_mix(const float* __restrict__ G, float* __restrict__ out_flat) {
  int id = blockIdx.x * blockDim.x + threadIdx.x;
  if (id >= B_TOT * 800) return;
  int b = id / 800; int r = id % 800;
  int o = r / 25; int pos = r % 25;
  int py = pos / 5, px = pos % 5;
  float m = -1e30f;
  #pragma unroll
  for (int dy = 0; dy < 2; dy++)
    #pragma unroll
    for (int dx = 0; dx < 2; dx++) {
      int p = (2 * py + dy) * 10 + (2 * px + dx);
      const float* gb = G + ((size_t)b * P_POS + p) * NPAD;
      float x1 = gb[o], x2 = gb[32 + o], gl = gb[64];
      float sg = 1.0f / (1.0f + __expf(-gl));
      float v = fmaf(x1 - x2, sg, x2);   // x1*g + x2*(1-g)
      v = fmaxf(v, 0.0f);
      m = fmaxf(m, v);
    }
  out_flat[id] = m;
}

// ---------------- Kernel 4: fc [B,800] x [10,800]^T + b
__global__ __launch_bounds__(256)
void k_fc(const float* __restrict__ flat, const float* __restrict__ fcw,
          const float* __restrict__ fcb, float* __restrict__ out) {
  int id = blockIdx.x * blockDim.x + threadIdx.x;
  if (id >= B_TOT * 10) return;
  int b = id / 10, j = id % 10;
  const float* fr = flat + (size_t)b * 800;
  const float* wr = fcw + j * 800;
  float s = fcb[j];
  for (int k = 0; k < 800; k++) s = fmaf(fr[k], wr[k], s);
  out[id] = s;
}

extern "C" void kernel_launch(void* const* d_in, const int* in_sizes, int n_in,
                              void* d_out, int out_size, void* d_ws, size_t ws_size,
                              hipStream_t stream) {
  const float* x   = (const float*)d_in[0];
  const float* c1w = (const float*)d_in[1];
  const float* c1b = (const float*)d_in[2];
  const float* ucw = (const float*)d_in[3];
  const float* ucb = (const float*)d_in[4];
  const float* pcw = (const float*)d_in[5];
  const float* pcb = (const float*)d_in[6];
  const float* wlw = (const float*)d_in[7];
  const float* wlb = (const float*)d_in[8];
  const float* fcw = (const float*)d_in[9];
  const float* fcb = (const float*)d_in[10];
  float* out = (float*)d_out;

  char* ws = (char*)d_ws;
  _Float16* h16   = (_Float16*)ws;                                  // 2048*196*16*2  = 12,845,056 B
  _Float16* wfrag = (_Float16*)(ws + 12845056);                     // 100*33280*2    =  6,656,000 B
  float*    G     = (float*)   (ws + 12845056 + 6656000);           // 2048*100*80*4  = 65,536,000 B

  k_conv1<<<(B_TOT * 196 + 255) / 256, 256, 0, stream>>>(x, c1w, c1b, h16);
  k_wswz<<<(P_POS * KT * NT * 512 + 255) / 256, 256, 0, stream>>>(ucw, ucb, pcw, pcb, wlw, wlb, wfrag);
  dim3 g2(B_TOT / MT, P_POS);
  k_gemm<<<g2, 256, 0, stream>>>(h16, wfrag, G);
  float* flat = out + B_TOT * 10;
  k_mix<<<(B_TOT * 800 + 255) / 256, 256, 0, stream>>>(G, flat);
  k_fc<<<(B_TOT * 10 + 255) / 256, 256, 0, stream>>>(flat, fcw, fcb, out);
}
